// GATC_cdk_82179904242302
// MI455X (gfx1250) — matrix operand order repaired and checked by reading, not compiled, not run on hardware
//
#include <hip/hip_runtime.h>
#include <math.h>

typedef __attribute__((ext_vector_type(16))) _Float16 v16h;
typedef __attribute__((ext_vector_type(8)))  float    v8f;

#define GN      50000
#define GE      800000
#define IN_CH   40
#define CDK     200
#define EDGE_DIM 10
#define HIDC    64
#define NEG_SLOPE 0.2f
#define SM_EPS  1e-16f
#define BN_EPS  1e-5f
#define HSTRIDE 256

__device__ __forceinline__ unsigned encf(float f) {
    unsigned u = __float_as_uint(f);
    return (u & 0x80000000u) ? ~u : (u | 0x80000000u);
}
__device__ __forceinline__ float decf(unsigned u) {
    return (u & 0x80000000u) ? __uint_as_float(u & 0x7fffffffu)
                             : __uint_as_float(~u);
}

// ---------------- h0 = concat(x, cdk), zero-padded to 256 cols ----------------
__global__ void k_concat(const float* __restrict__ x, const float* __restrict__ cdk,
                         float* __restrict__ h) {
    size_t idx = (size_t)blockIdx.x * blockDim.x + threadIdx.x;
    if (idx >= (size_t)GN * HSTRIDE) return;
    int c = (int)(idx & (HSTRIDE - 1));
    int n = (int)(idx >> 8);
    float v = 0.f;
    if (c < IN_CH)            v = x[(size_t)n * IN_CH + c];
    else if (c < IN_CH + CDK) v = cdk[(size_t)n * CDK + (c - IN_CH)];
    h[idx] = v;
}

// ---- pre-pack W (fp32 [K,Nout], zero-pad K to 256) into per-fragment f16 ----
// Bh layout: [ntile][ktile(8)][lane(32)][16 halves]; lane<16: K=kb+t (t<16),
// lane>=16: K=kb+16+t; col = ntile*16 + (lane&15).
__global__ void k_bpack(const float* __restrict__ W, _Float16* __restrict__ Bh,
                        int Nout, int Kdim) {
    int idx = blockIdx.x * blockDim.x + threadIdx.x;
    int total = (Nout >> 4) * 8 * 32 * 16;
    if (idx >= total) return;
    int t    = idx & 15;
    int lane = (idx >> 4) & 31;
    int kt   = (idx >> 9) & 7;
    int nt   = idx >> 12;
    int k    = kt * 32 + (lane >> 4) * 16 + t;
    int col  = nt * 16 + (lane & 15);
    float v  = (k < Kdim) ? W[(size_t)k * Nout + col] : 0.f;
    Bh[idx]  = (_Float16)v;
}

// ---------------- xp = h @ W via v_wmma_f32_16x16x32_f16 ----------------
__global__ void k_gemm_wmma(const float* __restrict__ A, const _Float16* __restrict__ Bh,
                            float* __restrict__ C, int Nout) {
    const int ntiles = Nout >> 4;
    const int totalTiles = (GN >> 4) * ntiles;
    int wid  = threadIdx.x >> 5;
    int lane = threadIdx.x & 31;
    int tile = blockIdx.x * (blockDim.x >> 5) + wid;
    if (tile >= totalTiles) return;          // wave-uniform: EXEC stays all-ones
    int mt = tile / ntiles, nt = tile % ntiles;
    int hi = lane >> 4;
    int r  = lane & 15;
    const float* arow = A + ((size_t)(mt * 16 + r)) * HSTRIDE;
    v8f acc = {};
#pragma unroll
    for (int kt = 0; kt < 8; ++kt) {
        const int kb = kt * 32;
        const float4* f4 = (const float4*)(arow + kb);
        float4 p0 = f4[hi * 4], p1 = f4[hi * 4 + 1];
        float4 p2 = f4[hi * 4 + 2], p3 = f4[hi * 4 + 3];
        v16h a;
#pragma unroll
        for (int j = 0; j < 4; ++j) {
            a[j]      = (_Float16)((&p0.x)[j]);
            a[4 + j]  = (_Float16)((&p1.x)[j]);
            a[8 + j]  = (_Float16)((&p2.x)[j]);
            a[12 + j] = (_Float16)((&p3.x)[j]);
        }
        v16h b = *(const v16h*)(Bh + (((size_t)(nt * 8 + kt)) * 32 + lane) * 16);
        acc = __builtin_amdgcn_wmma_f32_16x16x32_f16(
            false, a, false, b, (short)0, acc, false, false);
    }
#pragma unroll
    for (int j = 0; j < 8; ++j) {
        int mrow = mt * 16 + j + hi * 8;
        C[(size_t)mrow * Nout + nt * 16 + r] = acc[j];
    }
}

// ---------------- per-node attention logits al_s, al_d ----------------
__global__ void k_node_alpha(const float* __restrict__ xp, const float* __restrict__ as_,
                             const float* __restrict__ ad_, float* __restrict__ als,
                             float* __restrict__ ald, int H, int Nout) {
    int idx = blockIdx.x * blockDim.x + threadIdx.x;
    if (idx >= GN * H) return;
    int h = idx % H, n = idx / H;
    const float* xr = xp + (size_t)n * Nout + h * HIDC;
    float s = 0.f, d = 0.f;
    for (int c = 0; c < HIDC; ++c) {
        float v = xr[c];
        s += v * as_[h * HIDC + c];
        d += v * ad_[h * HIDC + c];
    }
    als[idx] = s; ald[idx] = d;
}

// ---------------- P[d][h] = sum_c We[d, h*64+c] * ae[h][c] ----------------
__global__ void k_pmat(const float* __restrict__ We, const float* __restrict__ ae,
                       float* __restrict__ P, int H, int Nout) {
    int idx = threadIdx.x;
    if (idx >= EDGE_DIM * H) return;
    int h = idx % H, d = idx / H;
    float s = 0.f;
    for (int c = 0; c < HIDC; ++c) s += We[(size_t)d * Nout + h * HIDC + c] * ae[h * HIDC + c];
    P[d * H + h] = s;
}

// ---- edge pass 1: alpha = lrelu(als[src]+ald[dst]+al_e); seg-max; sum(al_e) ----
__global__ void k_edge1(const int* __restrict__ src, const int* __restrict__ dst,
                        const float* __restrict__ ea, const float* __restrict__ P,
                        const float* __restrict__ als, const float* __restrict__ ald,
                        float* __restrict__ alph, unsigned* __restrict__ menc,
                        float* __restrict__ aleSum, int H) {
    __shared__ float ssum[4];
    int tid = threadIdx.x;
    if (tid < H) ssum[tid] = 0.f;
    __syncthreads();
    int e = blockIdx.x * blockDim.x + tid;
    float loc[4] = {0.f, 0.f, 0.f, 0.f};
    if (e < GE) {
        int s = src[e], d = dst[e];
        float attr[EDGE_DIM];
#pragma unroll
        for (int k = 0; k < EDGE_DIM; ++k) attr[k] = ea[(size_t)e * EDGE_DIM + k];
        for (int h = 0; h < H; ++h) {
            float ale = 0.f;
#pragma unroll
            for (int k = 0; k < EDGE_DIM; ++k) ale += attr[k] * P[k * H + h];
            loc[h] = ale;
            float a = als[(size_t)s * H + h] + ald[(size_t)d * H + h] + ale;
            a = (a > 0.f) ? a : NEG_SLOPE * a;
            alph[(size_t)e * H + h] = a;
            atomicMax(&menc[(size_t)d * H + h], encf(a));
        }
    }
    for (int h = 0; h < H; ++h) atomicAdd(&ssum[h], loc[h]);
    __syncthreads();
    if (tid < H) atomicAdd(&aleSum[tid], ssum[tid]);
}

// ---- self-loop: finalize max (m becomes float), init den & acc with self term ----
__global__ void k_selfinit(const float* __restrict__ als, const float* __restrict__ ald,
                           const float* __restrict__ aleSum, unsigned* __restrict__ menc,
                           float* __restrict__ den, const float* __restrict__ xp,
                           float* __restrict__ acc, int H, int Nout) {
    int idx = blockIdx.x * blockDim.x + threadIdx.x;
    if (idx >= GN * H) return;
    int h = idx % H, n = idx / H;
    float aself = als[idx] + ald[idx] + aleSum[h] * (1.f / (float)GE);
    aself = (aself > 0.f) ? aself : NEG_SLOPE * aself;
    unsigned ue = menc[idx];
    float m = (ue == 0u) ? aself : fmaxf(decf(ue), aself);
    ((float*)menc)[idx] = m;              // buffer now holds float max
    float exs = expf(aself - m);
    den[idx] = exs;
    const float* xr = xp + (size_t)n * Nout + h * HIDC;
    float* ar = acc + (size_t)n * Nout + h * HIDC;
    for (int c = 0; c < HIDC; ++c) ar[c] = exs * xr[c];
}

// ---- edge pass 2: ex = exp(alpha - m[dst]); den[dst] += ex ----
__global__ void k_edge2(const int* __restrict__ dst, float* __restrict__ alph,
                        const float* __restrict__ m, float* __restrict__ den, int H) {
    size_t idx = (size_t)blockIdx.x * blockDim.x + threadIdx.x;
    if (idx >= (size_t)GE * H) return;
    int h = (int)(idx % H);
    size_t e = idx / H;
    int d = dst[e];
    float ex = expf(alph[idx] - m[(size_t)d * H + h]);
    alph[idx] = ex;
    atomicAdd(&den[(size_t)d * H + h], ex);
}

// ---- edge pass 3: one wave per edge, acc[dst] += ex * xp[src] ----
__global__ void k_edge3(const int* __restrict__ src, const int* __restrict__ dst,
                        const float* __restrict__ alph, const float* __restrict__ xp,
                        float* __restrict__ acc, int H, int Nout) {
    int e = blockIdx.x * (blockDim.x >> 5) + (threadIdx.x >> 5);
    if (e >= GE) return;
    int lane = threadIdx.x & 31;
    int s = src[e], d = dst[e];
    const float* xr = xp + (size_t)s * Nout;
    float* ar = acc + (size_t)d * Nout;
    const float* al = alph + (size_t)e * H;
    for (int c = lane; c < Nout; c += 32) {
        atomicAdd(&ar[c], al[c >> 6] * xr[c]);
    }
}

// ---- batch-norm stats over normalized outputs (out = acc/(den+eps)) ----
__global__ void k_bn_stats(const float* __restrict__ acc, const float* __restrict__ den,
                           float* __restrict__ colsum, float* __restrict__ colsumsq,
                           int H, int Nout, int rowsPerBlock) {
    int c = threadIdx.x;            // blockDim.x == Nout
    int h = c >> 6;
    int n0 = blockIdx.x * rowsPerBlock;
    float s = 0.f, s2 = 0.f;
    for (int i = 0; i < rowsPerBlock; ++i) {
        int n = n0 + i;
        if (n < GN) {
            float v = acc[(size_t)n * Nout + c] / (den[(size_t)n * H + h] + SM_EPS);
            s += v; s2 += v * v;
        }
    }
    atomicAdd(&colsum[c], s);
    atomicAdd(&colsumsq[c], s2);
}

// ---- apply BN + ELU in place (acc aliases h-buffer for next layer) ----
__global__ void k_bn_elu(float* __restrict__ acc, const float* __restrict__ den,
                         const float* __restrict__ colsum, const float* __restrict__ colsumsq,
                         const float* __restrict__ g, const float* __restrict__ be,
                         int H, int Nout) {
    size_t idx = (size_t)blockIdx.x * blockDim.x + threadIdx.x;
    if (idx >= (size_t)GN * Nout) return;
    int c = (int)(idx % Nout);
    size_t n = idx / Nout;
    int h = c >> 6;
    float v  = acc[idx] / (den[n * H + h] + SM_EPS);
    float mu = colsum[c] * (1.f / (float)GN);
    float var = colsumsq[c] * (1.f / (float)GN) - mu * mu;
    float y = g[c] * (v - mu) * rsqrtf(var + BN_EPS) + be[c];
    acc[idx] = (y > 0.f) ? y : expm1f(y);
}

// ---- final: sigmoid(h2 @ linW + linB) ----
__global__ void k_final(const float* __restrict__ h2, const float* __restrict__ linW,
                        const float* __restrict__ linB, float* __restrict__ out) {
    int n = blockIdx.x * blockDim.x + threadIdx.x;
    if (n >= GN) return;
    float s = linB[0];
    for (int c = 0; c < HIDC; ++c) s += h2[(size_t)n * HIDC + c] * linW[c];
    out[n] = 1.f / (1.f + expf(-s));
}

extern "C" void kernel_launch(void* const* d_in, const int* in_sizes, int n_in,
                              void* d_out, int out_size, void* d_ws, size_t ws_size,
                              hipStream_t stream) {
    (void)in_sizes; (void)n_in; (void)out_size; (void)ws_size;
    const float* x    = (const float*)d_in[0];
    const float* cdk  = (const float*)d_in[1];
    const int*   esrc = (const int*)d_in[2];
    const int*   edst = (const int*)d_in[3];
    const float* ea   = (const float*)d_in[4];
    const float* linW = (const float*)d_in[29];
    const float* linB = (const float*)d_in[30];
    float* out = (float*)d_out;

    char* ws = (char*)d_ws;
    size_t off = 0;
    auto alloc = [&](size_t bytes) -> void* {
        void* p = ws + off;
        off += (bytes + 255) & ~(size_t)255;
        return p;
    };
    float*     hbuf = (float*)alloc((size_t)GN * HSTRIDE * 4);  // doubles as acc
    float*     xp   = (float*)alloc((size_t)GN * HSTRIDE * 4);
    float*     alph = (float*)alloc((size_t)GE * 4 * 4);
    float*     als  = (float*)alloc((size_t)GN * 4 * 4);
    float*     ald  = (float*)alloc((size_t)GN * 4 * 4);
    float*     mbuf = (float*)alloc((size_t)GN * 4 * 4);        // uint then float
    float*     den  = (float*)alloc((size_t)GN * 4 * 4);
    _Float16*  Bh   = (_Float16*)alloc((size_t)16 * 8 * 32 * 16 * 2);
    float*     smallb = (float*)alloc(4096);
    float* P       = smallb;          // 40
    float* aleSum  = smallb + 64;     // 4
    float* colsum  = smallb + 128;    // 256
    float* colsumsq = smallb + 512;   // 256

    // build h0
    {
        size_t tot = (size_t)GN * HSTRIDE;
        k_concat<<<(unsigned)((tot + 255) / 256), 256, 0, stream>>>(x, cdk, hbuf);
    }

    const int NoutA[3] = {256, 256, 64};
    const int KdimA[3] = {IN_CH + CDK, 256, 256};
    const int HA[3]    = {4, 4, 1};
    const int Wi[3]  = {5, 13, 21};
    const int asi[3] = {6, 14, 22};
    const int adi[3] = {7, 15, 23};
    const int Wei[3] = {8, 16, 24};
    const int aei[3] = {9, 17, 25};
    const int gi[3]  = {11, 19, 27};
    const int bei[3] = {12, 20, 28};

    for (int L = 0; L < 3; ++L) {
        const int Nout = NoutA[L], Kdim = KdimA[L], H = HA[L];
        const float* W   = (const float*)d_in[Wi[L]];
        const float* as_ = (const float*)d_in[asi[L]];
        const float* ad_ = (const float*)d_in[adi[L]];
        const float* We  = (const float*)d_in[Wei[L]];
        const float* ae  = (const float*)d_in[aei[L]];
        const float* g   = (const float*)d_in[gi[L]];
        const float* be  = (const float*)d_in[bei[L]];

        hipMemsetAsync(mbuf, 0, (size_t)GN * H * 4, stream);   // encoded -inf
        hipMemsetAsync(smallb, 0, 4096, stream);               // P/aleSum/col stats

        // pack weights to f16 fragments
        {
            int total = (Nout >> 4) * 8 * 32 * 16;
            k_bpack<<<(total + 255) / 256, 256, 0, stream>>>(W, Bh, Nout, Kdim);
        }
        // xp = h @ W  (WMMA)
        {
            int tiles = (GN >> 4) * (Nout >> 4);
            k_gemm_wmma<<<(tiles + 7) / 8, 256, 0, stream>>>(hbuf, Bh, xp, Nout);
        }
        // node logits
        k_node_alpha<<<(GN * H + 255) / 256, 256, 0, stream>>>(xp, as_, ad_, als, ald, H, Nout);
        // P = We . ae
        k_pmat<<<1, 64, 0, stream>>>(We, ae, P, H, Nout);
        // edge pass 1
        k_edge1<<<(GE + 255) / 256, 256, 0, stream>>>(esrc, edst, ea, P, als, ald,
                                                      alph, (unsigned*)mbuf, aleSum, H);
        // self-loop init (also seeds acc = hbuf, overwriting consumed h)
        k_selfinit<<<(GN * H + 255) / 256, 256, 0, stream>>>(als, ald, aleSum,
                                                             (unsigned*)mbuf, den, xp, hbuf, H, Nout);
        // edge pass 2 (denominator)
        {
            size_t tot = (size_t)GE * H;
            k_edge2<<<(unsigned)((tot + 255) / 256), 256, 0, stream>>>(edst, alph, mbuf, den, H);
        }
        // edge pass 3 (aggregation)
        k_edge3<<<(GE + 7) / 8, 256, 0, stream>>>(esrc, edst, alph, xp, hbuf, H, Nout);
        // batch-norm stats + apply + ELU (in place -> next layer input)
        k_bn_stats<<<(GN + 255) / 256, Nout, 0, stream>>>(hbuf, den, colsum, colsumsq, H, Nout, 256);
        {
            size_t tot = (size_t)GN * Nout;
            k_bn_elu<<<(unsigned)((tot + 255) / 256), 256, 0, stream>>>(hbuf, den, colsum,
                                                                        colsumsq, g, be, H, Nout);
        }
    }

    // final linear + sigmoid (layer-2 output is stride-64 in hbuf)
    k_final<<<(GN + 255) / 256, 256, 0, stream>>>(hbuf, linW, linB, out);
}
